// RPCAttention_10943576670640
// MI455X (gfx1250) — matrix operand order repaired and checked by reading, not compiled, not run on hardware
//
#include <hip/hip_runtime.h>
#include <hip/hip_bf16.h>
#include <math.h>

// ---------------- constants (match reference) ----------------
constexpr int BATCH   = 8;
constexpr int SEQ     = 512;
constexpr int DIM     = 768;
constexpr int HEADS   = 12;
constexpr int HD      = 64;            // head dim
constexpr int THREE_D = 3 * DIM;       // 2304
constexpr int BH      = BATCH * HEADS; // 96
constexpr int RANK    = 64;            // subspace rank for SVT
constexpr float SCALE = 0.125f;        // 1/sqrt(64)
constexpr float LAMBDA_SPARSE = 0.1f;
constexpr float SVD_TAU = 1.0f;
constexpr int PCP_ITERS = 10;
constexpr int SUBSPACE_ITERS = 2;

typedef __attribute__((ext_vector_type(16))) _Float16 v16h;
typedef __attribute__((ext_vector_type(8)))  _Float16 h8;
typedef __attribute__((ext_vector_type(8)))  float    v8f;

// ---------------- f32 -> f16 conversion helpers (vectorized loads) ----------------
__device__ __forceinline__ h8 cvt8(const float* __restrict__ src) {
  const float4 f0 = *(const float4*)(src);
  const float4 f1 = *(const float4*)(src + 4);
  h8 r;
  r[0] = (_Float16)f0.x; r[1] = (_Float16)f0.y; r[2] = (_Float16)f0.z; r[3] = (_Float16)f0.w;
  r[4] = (_Float16)f1.x; r[5] = (_Float16)f1.y; r[6] = (_Float16)f1.z; r[7] = (_Float16)f1.w;
  return r;
}
__device__ __forceinline__ h8 cvt8_sub(const float* __restrict__ a,
                                       const float* __restrict__ s) {
  const float4 a0 = *(const float4*)(a);
  const float4 a1 = *(const float4*)(a + 4);
  const float4 s0 = *(const float4*)(s);
  const float4 s1 = *(const float4*)(s + 4);
  h8 r;
  r[0] = (_Float16)(a0.x - s0.x); r[1] = (_Float16)(a0.y - s0.y);
  r[2] = (_Float16)(a0.z - s0.z); r[3] = (_Float16)(a0.w - s0.w);
  r[4] = (_Float16)(a1.x - s1.x); r[5] = (_Float16)(a1.y - s1.y);
  r[6] = (_Float16)(a1.z - s1.z); r[7] = (_Float16)(a1.w - s1.w);
  return r;
}
__device__ __forceinline__ void cvt16(const float* __restrict__ src, _Float16* dst) {
  #pragma unroll
  for (int i = 0; i < 4; ++i) {
    const float4 f = *(const float4*)(src + 4 * i);
    dst[4*i+0] = (_Float16)f.x; dst[4*i+1] = (_Float16)f.y;
    dst[4*i+2] = (_Float16)f.z; dst[4*i+3] = (_Float16)f.w;
  }
}
__device__ __forceinline__ void cvt16_sub(const float* __restrict__ a,
                                          const float* __restrict__ s, _Float16* dst) {
  #pragma unroll
  for (int i = 0; i < 4; ++i) {
    const float4 fa = *(const float4*)(a + 4 * i);
    const float4 fs = *(const float4*)(s + 4 * i);
    dst[4*i+0] = (_Float16)(fa.x - fs.x); dst[4*i+1] = (_Float16)(fa.y - fs.y);
    dst[4*i+2] = (_Float16)(fa.z - fs.z); dst[4*i+3] = (_Float16)(fa.w - fs.w);
  }
}

// ---------------- generic strided-batched WMMA GEMM ----------------
// C[m,n] = alpha * sum_k op(A)[m,k] * op(B)[k,n]
// SUB: op(A) reads (A - Asub) elementwise (fuses M = A - S), branch-free.
// Block = 256 threads = 8 wave32s; block tile 32(M) x 64(N); each wave owns one
// 16x16 C tile, marching K in steps of 32 with v_wmma_f32_16x16x32_f16.
// All global loads are b128 (lane's 16 staged elements are contiguous per case);
// LDS tiles laid out so fragment gathers are 2x ds_load_b128 per operand.
struct GemmParams {
  const float* A; const float* Asub; const float* B;
  float* C;
  long lda, ldb, ldc;
  long sA1, sA2, sB1, sB2, sC1, sC2;  // off = (z/batch2)*s1 + (z%batch2)*s2
  int M, N, K, batch2;
  float alpha;
};

template<bool TA, bool TB, bool SUB>
__global__ __launch_bounds__(256)
void wmma_gemm(GemmParams p) {
  __shared__ _Float16 lA [8][16][32];   // per-wave op(A) tile: [m][k]
  __shared__ _Float16 lBT[8][16][32];   // per-wave op(B) tile TRANSPOSED: [n][k]

  const int tid  = threadIdx.x;
  const int wave = tid >> 5;
  const int lane = tid & 31;

  const int z = blockIdx.z;
  const long offA = (long)(z / p.batch2) * p.sA1 + (long)(z % p.batch2) * p.sA2;
  const long offB = (long)(z / p.batch2) * p.sB1 + (long)(z % p.batch2) * p.sB2;
  const long offC = (long)(z / p.batch2) * p.sC1 + (long)(z % p.batch2) * p.sC2;

  const int tileM = blockIdx.y * 32 + (wave >> 2) * 16;
  const int tileN = blockIdx.x * 64 + (wave & 3) * 16;

  // staging roles
  const int ar  = lane >> 1;          // !TA: A tile row 0..15
  const int ac0 = (lane & 1) * 16;    // !TA: A tile k segment
  const int btn = lane & 15;          //  TB: B tile col (n)
  const int bts = (lane >> 4) * 16;   //  TB: B tile k segment
  // fragment roles (ISA 7.12.2 layouts, wave32)
  const int am  = lane & 15;          // A frag row
  const int akb = (lane >> 4) * 8;    // A frag K base (0 or 8)
  const int bn  = lane & 15;          // B frag col
  const int bkb = (lane >> 4) * 8;    // B frag K base (0 or 8)

  v8f acc = {};

  for (int k0 = 0; k0 < p.K; k0 += 32) {
    // ---- stage op(A) tile (16m x 32k) as f16 ----
    if constexpr (!TA) {
      const long base = offA + (long)(tileM + ar) * p.lda + k0 + ac0;
      h8* dst = (h8*)&lA[wave][ar][ac0];
      if constexpr (SUB) {
        dst[0] = cvt8_sub(p.A + base, p.Asub + base);
        dst[1] = cvt8_sub(p.A + base + 8, p.Asub + base + 8);
      } else {
        dst[0] = cvt8(p.A + base);
        dst[1] = cvt8(p.A + base + 8);
      }
      if (k0 + 32 < p.K) __builtin_prefetch(p.A + base + 32, 0, 3);
    } else {
      // lane = k row (0..31); 16 contiguous m
      const long base = offA + (long)(k0 + lane) * p.lda + tileM;
      _Float16 hh[16];
      if constexpr (SUB) cvt16_sub(p.A + base, p.Asub + base, hh);
      else               cvt16(p.A + base, hh);
      #pragma unroll
      for (int i = 0; i < 16; ++i) lA[wave][i][lane] = hh[i];
      if (k0 + 32 < p.K) __builtin_prefetch(p.A + base + 32 * p.lda, 0, 3);
    }
    // ---- stage op(B) tile (32k x 16n) as f16, transposed into lBT[n][k] ----
    if constexpr (!TB) {
      // lane = k row (0..31); 16 contiguous n
      const long base = offB + (long)(k0 + lane) * p.ldb + tileN;
      _Float16 hh[16];
      cvt16(p.B + base, hh);
      #pragma unroll
      for (int c = 0; c < 16; ++c) lBT[wave][c][lane] = hh[c];
      if (k0 + 32 < p.K) __builtin_prefetch(p.B + base + 32 * p.ldb, 0, 3);
    } else {
      // lane: n = lane&15, k segment = (lane>>4)*16; 16 contiguous k
      const long base = offB + (long)(tileN + btn) * p.ldb + k0 + bts;
      h8* dst = (h8*)&lBT[wave][btn][bts];
      dst[0] = cvt8(p.B + base);
      dst[1] = cvt8(p.B + base + 8);
      if (k0 + 32 < p.K) __builtin_prefetch(p.B + base + 32, 0, 3);
    }
    // same-wave LDS is in-order; drain DScnt before fragment reads
    asm volatile("s_wait_dscnt 0" ::: "memory");

    union U16 { v16h v; h8 h[2]; };
    U16 ua, ub;
    ua.h[0] = *(const h8*)&lA [wave][am][akb];        // K = akb + 0..7
    ua.h[1] = *(const h8*)&lA [wave][am][16 + akb];   // K = 16 + akb + 0..7
    ub.h[0] = *(const h8*)&lBT[wave][bn][bkb];        // K = bkb + 0..7
    ub.h[1] = *(const h8*)&lBT[wave][bn][16 + bkb];   // K = 16 + bkb + 0..7

    acc = __builtin_amdgcn_wmma_f32_16x16x32_f16(
        /*neg_a=*/false, ua.v, /*neg_b=*/false, ub.v,
        /*c_mod=*/(short)0, acc, /*reuse_a=*/false, /*reuse_b=*/false);
  }

  // epilogue: C layout — VGPR r holds M = r + 8*(lane>>4), N = lane&15
  #pragma unroll
  for (int r = 0; r < 8; ++r) {
    const int m = tileM + r + (lane >> 4) * 8;
    const int n = tileN + (lane & 15);
    p.C[offC + (long)m * p.ldc + n] = p.alpha * acc[r];
  }
}

// ---------------- rowwise softmax of (in1 [+ in2]), 512 cols, wave/row ----------------
__global__ __launch_bounds__(256)
void softmax_rows(const float* in1, const float* in2, float* out, int rows) {
  const int wave = threadIdx.x >> 5, lane = threadIdx.x & 31;
  const int row = blockIdx.x * 8 + wave;
  if (row >= rows) return;
  const long base = (long)row * SEQ;
  float v[16];
  float mx = -1e30f;
  #pragma unroll
  for (int i = 0; i < 16; ++i) {
    float x = in1[base + lane + i * 32];
    if (in2) x += in2[base + lane + i * 32];
    v[i] = x;
    mx = fmaxf(mx, x);
  }
  #pragma unroll
  for (int m = 16; m >= 1; m >>= 1) mx = fmaxf(mx, __shfl_xor(mx, m, 32));
  float s = 0.f;
  #pragma unroll
  for (int i = 0; i < 16; ++i) { v[i] = __expf(v[i] - mx); s += v[i]; }
  #pragma unroll
  for (int m = 16; m >= 1; m >>= 1) s += __shfl_xor(s, m, 32);
  const float inv = 1.f / s;
  #pragma unroll
  for (int i = 0; i < 16; ++i) out[base + lane + i * 32] = v[i] * inv;
}

// ------- per-batch: chol(Gram) then V = T * R^-1 (orthonormalize columns) -------
__global__ __launch_bounds__(256)
void orth_kernel(const float* Gram, const float* T, float* V) {
  __shared__ float R[RANK][RANK + 1];
  const int b = blockIdx.x;
  const int tid = threadIdx.x;
  for (int i = tid; i < RANK * RANK; i += 256)
    R[i >> 6][i & 63] = Gram[(long)b * RANK * RANK + i];
  __syncthreads();
  if (tid < RANK) R[tid][tid] += 1e-4f;   // ridge for PSD safety
  __syncthreads();

  if (tid < 32) {                          // one wave32 does the 64x64 Cholesky
    const int lane = tid;
    for (int k = 0; k < RANK; ++k) {
      const float d = sqrtf(fmaxf(R[k][k], 1e-12f));
      for (int j = k + lane; j < RANK; j += 32) R[k][j] = R[k][j] / d;
      for (int i = k + 1; i < RANK; ++i)
        for (int j = i + lane; j < RANK; j += 32) R[i][j] -= R[k][i] * R[k][j];
    }
  }
  __syncthreads();

  // forward substitution: v * R = t (rows of T); full unroll keeps v[] in VGPRs
  for (int r = tid; r < SEQ; r += 256) {
    const long rowb = (long)b * SEQ * RANK + (long)r * RANK;
    float v[RANK];
    #pragma unroll
    for (int j = 0; j < RANK; ++j) {
      float s = T[rowb + j];
      #pragma unroll
      for (int i = 0; i < j; ++i) s -= v[i] * R[i][j];
      v[j] = s / R[j][j];
    }
    #pragma unroll
    for (int j = 0; j < RANK; ++j) V[rowb + j] = v[j];
  }
}

// ---- per-batch: sigma_j = ||Y[:,j]||, then Y[:,j] *= max(1 - tau/sigma, 0) ----
__global__ __launch_bounds__(256)
void colnorm_scale(float* Y, float tau) {
  __shared__ float part[4][RANK];
  __shared__ float g[RANK];
  const int b = blockIdx.x;
  const long base = (long)b * SEQ * RANK;
  const int col = threadIdx.x & 63;
  const int seg = threadIdx.x >> 6;            // 0..3, 128 rows each
  float s = 0.f;
  for (int r = seg * 128; r < seg * 128 + 128; ++r) {
    const float x = Y[base + (long)r * RANK + col];
    s += x * x;
  }
  part[seg][col] = s;
  __syncthreads();
  if (threadIdx.x < RANK) {
    const float t = part[0][threadIdx.x] + part[1][threadIdx.x] +
                    part[2][threadIdx.x] + part[3][threadIdx.x];
    const float sig = sqrtf(t);
    g[threadIdx.x] = (sig > tau) ? (1.f - tau / sig) : 0.f;
  }
  __syncthreads();
  for (int i = threadIdx.x; i < SEQ * RANK; i += 256)
    Y[base + i] *= g[i & 63];
}

// ---------------- S = soft_threshold(A - L, lam) ----------------
__global__ __launch_bounds__(256)
void softthresh(const float* A, const float* L, float* S, float lam, long n) {
  for (long i = (long)blockIdx.x * 256 + threadIdx.x; i < n;
       i += (long)gridDim.x * 256) {
    const float d = A[i] - L[i];
    const float m = fabsf(d) - lam;
    S[i] = (m > 0.f) ? copysignf(m, d) : 0.f;
  }
}

// ---------------- deterministic pseudo-random init of V ----------------
__global__ __launch_bounds__(256)
void init_V(float* V, long n) {
  for (long i = (long)blockIdx.x * 256 + threadIdx.x; i < n;
       i += (long)gridDim.x * 256) {
    unsigned h = (unsigned)i * 2654435761u;
    h ^= h >> 16; h *= 2246822519u; h ^= h >> 13;
    V[i] = ((float)(h & 0xFFFF) * (1.f / 65536.f)) - 0.5f;
  }
}

// ---------------- host-side helpers ----------------
static inline void launch_gemm(hipStream_t st, bool ta, bool tb, int batches,
                               const GemmParams& p) {
  dim3 g(p.N / 64, p.M / 32, batches), b(256);
  const bool sub = (p.Asub != nullptr);
  if (!ta && !tb && !sub)      wmma_gemm<false, false, false><<<g, b, 0, st>>>(p);
  else if (!ta && !tb &&  sub) wmma_gemm<false, false, true ><<<g, b, 0, st>>>(p);
  else if ( ta && !tb &&  sub) wmma_gemm<true,  false, true ><<<g, b, 0, st>>>(p);
  else if ( ta && !tb && !sub) wmma_gemm<true,  false, false><<<g, b, 0, st>>>(p);
  else                         wmma_gemm<false, true,  false><<<g, b, 0, st>>>(p);
}

extern "C" void kernel_launch(void* const* d_in, const int* in_sizes, int n_in,
                              void* d_out, int out_size, void* d_ws, size_t ws_size,
                              hipStream_t stream) {
  const float* x    = (const float*)d_in[0];  // [B,S,768]
  const float* Wqkv = (const float*)d_in[1];  // [768,2304]
  const float* Wout = (const float*)d_in[2];  // [768,768]
  float* out = (float*)d_out;                 // [B,S,768]

  // workspace carve-up (f32)
  float* p = (float*)d_ws;
  float* qkv = p; p += (long)BATCH * SEQ * THREE_D;   // [B,S,2304]
  float* Aat = p; p += (long)BH * SEQ * SEQ;          // A (then final attn)
  float* Lm  = p; p += (long)BH * SEQ * SEQ;          // L
  float* Sm  = p; p += (long)BH * SEQ * SEQ;          // S (sparse)
  float* V   = p; p += (long)BH * SEQ * RANK;         // right-subspace basis
  float* Y   = p; p += (long)BH * SEQ * RANK;         // Y = M V
  float* T   = p; p += (long)BH * SEQ * RANK;         // T = M^T Y
  float* Gr  = p; p += (long)BH * RANK * RANK;        // Gram = T^T T
  float* ctx = p; p += (long)BATCH * SEQ * DIM;       // attn @ v, [B,S,768]

  const long NAS = (long)BH * SEQ * SEQ;

  // 1) qkv = x @ W_qkv                    [4096x768]@[768x2304]
  { GemmParams g{ x, nullptr, Wqkv, qkv, DIM, THREE_D, THREE_D,
                  0,0, 0,0, 0,0, BATCH*SEQ, THREE_D, DIM, 1, 1.0f };
    launch_gemm(stream, false, false, 1, g); }

  // 2) scores[b,h] = (q @ k^T) * SCALE    per-(b,h) 512x512, K=64
  { GemmParams g{ qkv, nullptr, qkv + DIM, Aat, THREE_D, THREE_D, SEQ,
                  (long)SEQ*THREE_D, HD, (long)SEQ*THREE_D, HD,
                  (long)HEADS*SEQ*SEQ, (long)SEQ*SEQ,
                  SEQ, SEQ, HD, HEADS, SCALE };
    launch_gemm(stream, false, true, BH, g); }

  // 3) A = softmax(scores) (in place)
  softmax_rows<<<dim3(BH * SEQ / 8), dim3(256), 0, stream>>>(Aat, nullptr, Aat, BH * SEQ);

  // 4) S = 0, V = deterministic pseudo-random
  hipMemsetAsync(Sm, 0, NAS * sizeof(float), stream);
  init_V<<<dim3(4096), dim3(256), 0, stream>>>(V, (long)BH * SEQ * RANK);

  // 5) PCP loop: L = SVT(A - S) via rank-64 implicit subspace iteration
  for (int it = 0; it < PCP_ITERS; ++it) {
    for (int si = 0; si < SUBSPACE_ITERS; ++si) {
      // Y = (A - S) @ V
      { GemmParams g{ Aat, Sm, V, Y, SEQ, RANK, RANK,
                      (long)SEQ*SEQ, 0, (long)SEQ*RANK, 0, (long)SEQ*RANK, 0,
                      SEQ, RANK, SEQ, 1, 1.0f };
        launch_gemm(stream, false, false, BH, g); }
      // T = (A - S)^T @ Y
      { GemmParams g{ Aat, Sm, Y, T, SEQ, RANK, RANK,
                      (long)SEQ*SEQ, 0, (long)SEQ*RANK, 0, (long)SEQ*RANK, 0,
                      SEQ, RANK, SEQ, 1, 1.0f };
        launch_gemm(stream, true, false, BH, g); }
      // Gram = T^T @ T
      { GemmParams g{ T, nullptr, T, Gr, RANK, RANK, RANK,
                      (long)SEQ*RANK, 0, (long)SEQ*RANK, 0, (long)RANK*RANK, 0,
                      RANK, RANK, SEQ, 1, 1.0f };
        launch_gemm(stream, true, false, BH, g); }
      // V = T @ chol(Gram)^-1
      orth_kernel<<<dim3(BH), dim3(256), 0, stream>>>(Gr, T, V);
    }
    // Y = (A - S) @ V   (Y columns are sigma_j * u_j)
    { GemmParams g{ Aat, Sm, V, Y, SEQ, RANK, RANK,
                    (long)SEQ*SEQ, 0, (long)SEQ*RANK, 0, (long)SEQ*RANK, 0,
                    SEQ, RANK, SEQ, 1, 1.0f };
      launch_gemm(stream, false, false, BH, g); }
    // Y[:,j] *= max(1 - tau/sigma_j, 0)
    colnorm_scale<<<dim3(BH), dim3(256), 0, stream>>>(Y, SVD_TAU);
    // L = Y @ V^T
    { GemmParams g{ Y, nullptr, V, Lm, RANK, RANK, SEQ,
                    (long)SEQ*RANK, 0, (long)SEQ*RANK, 0, (long)SEQ*SEQ, 0,
                    SEQ, SEQ, RANK, 1, 1.0f };
      launch_gemm(stream, false, true, BH, g); }
    // S = soft_threshold(A - L, lambda)
    softthresh<<<dim3(8192), dim3(256), 0, stream>>>(Aat, Lm, Sm, LAMBDA_SPARSE, NAS);
  }

  // 6) attn = softmax(L + S)  -> reuse Aat
  softmax_rows<<<dim3(BH * SEQ / 8), dim3(256), 0, stream>>>(Lm, Sm, Aat, BH * SEQ);

  // 7) ctx[b,s,h*64+d] = attn[b,h] @ v[b,h]   per-(b,h) 512x64, K=512
  { GemmParams g{ Aat, nullptr, qkv + 2 * DIM, ctx, SEQ, THREE_D, DIM,
                  (long)HEADS*SEQ*SEQ, (long)SEQ*SEQ,
                  (long)SEQ*THREE_D, HD,
                  (long)SEQ*DIM, HD,
                  SEQ, HD, SEQ, HEADS, 1.0f };
    launch_gemm(stream, false, false, BH, g); }

  // 8) out = ctx @ W_out                  [4096x768]@[768x768]
  { GemmParams g{ ctx, nullptr, Wout, out, DIM, DIM, DIM,
                  0,0, 0,0, 0,0, BATCH*SEQ, DIM, DIM, 1, 1.0f };
    launch_gemm(stream, false, false, 1, g); }
}
